// FlowPyramid_43593918055090
// MI455X (gfx1250) — matrix operand order repaired and checked by reading, not compiled, not run on hardware
//
#include <hip/hip_runtime.h>

// ---------------------------------------------------------------------------
// PWC-Net forward for MI455X (gfx1250, wave32).
// All 3x3 convs run as implicit-GEMM on v_wmma_f32_16x16x32_bf16:
//   activations stay fp32 in memory (L2-resident, 192MB), A/B fragments are
//   bf16, accumulation fp32. Weights are repacked per conv into the contiguous-k
//   B-fragment layout (zero-padded to a multiple of 4 N-tiles) so the WMMA
//   loop is branch-free with EXEC all-ones, and each lane loads its B
//   fragment with 2x global_load_b128.
// ---------------------------------------------------------------------------

typedef __attribute__((ext_vector_type(16))) __bf16 v16bf;
typedef __attribute__((ext_vector_type(8)))  float  v8f;

union AFrag { v16bf v; unsigned short u[16]; };

static __device__ __forceinline__ unsigned short f2bf(float f) {
  unsigned u = __float_as_uint(f);
  unsigned r = u + 0x7FFFu + ((u >> 16) & 1u);   // round-to-nearest-even
  return (unsigned short)(r >> 16);
}

// -------------------- weight pack: HWIO f32 -> bf16 B-fragments -------------
// Layout: packed[(((nt*Ktiles + kt)*32 + lane)*16 + j]
//   lane: N = nt*16 + (lane&15);  K = kt*32 + (lane>>4)*16 + j   (j = 0..15)
// NtilesPad >= ceil(Cout/16) rounded up to NTPW; pad region is zero-filled.
__global__ void pack_w_k(const float* __restrict__ w, unsigned short* __restrict__ out,
                         int K, int Cout, int Ktiles, int NtilesPad) {
  size_t total = (size_t)NtilesPad * Ktiles * 512;
  for (size_t t = (size_t)blockIdx.x * blockDim.x + threadIdx.x; t < total;
       t += (size_t)gridDim.x * blockDim.x) {
    int j    = (int)(t & 15);
    int lane = (int)((t >> 4) & 31);
    size_t tile = t >> 9;
    int kt = (int)(tile % Ktiles);
    int nt = (int)(tile / Ktiles);
    int k = kt * 32 + (lane >> 4) * 16 + j;
    int n = nt * 16 + (lane & 15);
    float v = (k < K && n < Cout) ? w[(size_t)k * Cout + n] : 0.0f;
    out[t] = f2bf(v);
  }
}

// -------------------- implicit-GEMM 3x3 conv (SAME, dilated) ----------------
// Per wave: 16 spatial positions (M) x 4 cout tiles (N=64, zero-padded).
// K = 9*CinTot, tap-major (matches HWIO weight layout). Up to 5 concat srcs.
#define NTPW 4
__global__ __launch_bounds__(256)
void conv3x3_wmma_k(const float* __restrict__ s0, int c0,
                    const float* __restrict__ s1, int c1,
                    const float* __restrict__ s2, int c2,
                    const float* __restrict__ s3, int c3,
                    const float* __restrict__ s4, int c4,
                    const unsigned short* __restrict__ wpk,
                    const float* __restrict__ bias,
                    float* __restrict__ out,
                    int NB, int H, int W, int Cout, int dil, int leaky,
                    int CinTot, int Ktot, int Ktiles, int ngrp, int Mtiles) {
  const int P = NB * H * W;
  const int totalWaves = Mtiles * ngrp;
  // Force wave-uniform id into an SGPR so the guard is a scalar branch and
  // EXEC stays all-ones through the WMMA loop (ISA 7.12 requirement).
  int wid = __builtin_amdgcn_readfirstlane(
      (int)((blockIdx.x * blockDim.x + threadIdx.x) >> 5));
  if (wid >= totalWaves) return;

  const int lane = threadIdx.x & 31;
  const int grp  = wid % ngrp;
  const int mt   = wid / ngrp;
  const int nt0  = grp * NTPW;
  const int half = lane >> 4;
  const int mloc = lane & 15;
  const int m = mt * 16 + mloc;
  const bool mvalid = (m < P);
  int b = 0, y = 0, x = 0;
  if (mvalid) {
    b = m / (H * W);
    int r = m - b * H * W;
    y = r / W;
    x = r - y * W;
  }

  v8f acc0, acc1, acc2, acc3;
  for (int i = 0; i < 8; i++) { acc0[i] = 0.0f; acc1[i] = 0.0f; acc2[i] = 0.0f; acc3[i] = 0.0f; }

  const size_t laneB = (size_t)lane * 16;

  for (int kt = 0; kt < Ktiles; kt++) {
    AFrag a;
    // Two runs of 8 consecutive k per lane (contiguous k = half*16 + s, matches packed B).
    for (int run = 0; run < 2; run++) {
      int kstart = kt * 32 + half * 16 + run * 8;
      unsigned tap = (unsigned)kstart / (unsigned)CinTot;
      int ci = kstart - (int)tap * CinTot;
      int dy = (int)(tap / 3u) - 1, dx = (int)(tap % 3u) - 1;
      int iy = y + dy * dil, ix = x + dx * dil;
      for (int j = 0; j < 8; j++) {
        int k = kstart + j;
        float val = 0.0f;
        if (mvalid && k < Ktot && iy >= 0 && iy < H && ix >= 0 && ix < W) {
          // resolve concat segment (unrolled if-chain, no private arrays)
          int cc = ci;
          const float* s; int cs;
          if (cc < c0) { s = s0; cs = c0; }
          else { cc -= c0;
            if (cc < c1) { s = s1; cs = c1; }
            else { cc -= c1;
              if (cc < c2) { s = s2; cs = c2; }
              else { cc -= c2;
                if (cc < c3) { s = s3; cs = c3; }
                else { cc -= c3; s = s4; cs = c4; } } } }
          val = s[(((size_t)b * H + iy) * W + ix) * cs + cc];
        }
        a.u[run * 8 + j] = f2bf(val);
        // advance channel / tap
        ci++;
        if (ci >= CinTot) {
          ci = 0; tap++;
          dy = (int)(tap / 3u) - 1; dx = (int)(tap % 3u) - 1;
          iy = y + dy * dil; ix = x + dx * dil;
        }
      }
    }
    // 4 unconditional WMMAs reuse the same A fragment (B is zero-padded).
    const unsigned short* wb = wpk + (size_t)kt * 512 + laneB;
    const size_t tileStride = (size_t)Ktiles * 512;
    v16bf b0 = *(const v16bf*)(wb + (size_t)(nt0 + 0) * tileStride);
    v16bf b1 = *(const v16bf*)(wb + (size_t)(nt0 + 1) * tileStride);
    v16bf b2 = *(const v16bf*)(wb + (size_t)(nt0 + 2) * tileStride);
    v16bf b3 = *(const v16bf*)(wb + (size_t)(nt0 + 3) * tileStride);
    acc0 = __builtin_amdgcn_wmma_f32_16x16x32_bf16(false, a.v, false, b0, (short)0, acc0, false, false);
    acc1 = __builtin_amdgcn_wmma_f32_16x16x32_bf16(false, a.v, false, b1, (short)0, acc1, false, false);
    acc2 = __builtin_amdgcn_wmma_f32_16x16x32_bf16(false, a.v, false, b2, (short)0, acc2, false, false);
    acc3 = __builtin_amdgcn_wmma_f32_16x16x32_bf16(false, a.v, false, b3, (short)0, acc3, false, false);
  }

  // epilogue: bias + leaky-relu + scatter store (C/D layout: M=half*8+i, N=lane&15)
  v8f accs[NTPW] = {acc0, acc1, acc2, acc3};
  for (int t = 0; t < NTPW; t++) {
    int n = (nt0 + t) * 16 + mloc;
    if (n >= Cout) continue;          // padded N-tiles: nothing to store
    float bv = bias[n];
    for (int i = 0; i < 8; i++) {
      int mo = mt * 16 + half * 8 + i;
      if (mo < P) {
        float v = accs[t][i] + bv;
        if (leaky) v = (v > 0.0f) ? v : 0.1f * v;
        out[(size_t)mo * Cout + n] = v;
      }
    }
  }
}

// -------------------- cost volume (81 offsets, mean over C, leaky) ----------
__global__ void costvol_k(const float* __restrict__ f1, const float* __restrict__ f2,
                          float* __restrict__ out, int NB, int H, int W, int C) {
  int n = NB * H * W * 81;
  int tid = blockIdx.x * blockDim.x + threadIdx.x;
  if (tid >= n) return;
  int o = tid % 81;
  int p = tid / 81;
  int x = p % W, y = (p / W) % H, b = p / (W * H);
  int dy = o / 9 - 4, dx = o % 9 - 4;
  int yy = y + dy, xx = x + dx;
  float s = 0.0f;
  if (yy >= 0 && yy < H && xx >= 0 && xx < W) {
    const float* a  = f1 + (size_t)p * C;
    const float* bb = f2 + (size_t)((b * H + yy) * W + xx) * C;
    for (int c = 0; c < C; c++) s += a[c] * bb[c];
    s /= (float)C;
  }
  out[tid] = (s > 0.0f) ? s : 0.1f * s;
}

// -------------------- bilinear warp ----------------------------------------
__global__ void warp_k(const float* __restrict__ img, const float* __restrict__ flow,
                       float* __restrict__ out, int NB, int H, int W, int C) {
  int n = NB * H * W * C;
  int tid = blockIdx.x * blockDim.x + threadIdx.x;
  if (tid >= n) return;
  int c = tid % C;
  int p = tid / C;
  int x = p % W, y = (p / W) % H, b = p / (W * H);
  float qy = (float)y + flow[(size_t)p * 2 + 0];
  float qx = (float)x + flow[(size_t)p * 2 + 1];
  float fy0 = floorf(qy); fy0 = fminf(fmaxf(fy0, 0.0f), (float)(H - 2));
  float fx0 = floorf(qx); fx0 = fminf(fmaxf(fx0, 0.0f), (float)(W - 2));
  int y0 = (int)fy0, x0 = (int)fx0;
  float ay = fminf(fmaxf(qy - fy0, 0.0f), 1.0f);
  float ax = fminf(fmaxf(qx - fx0, 0.0f), 1.0f);
  size_t base = ((size_t)(b * H + y0) * W + x0) * C + c;
  float tl = img[base];
  float tr = img[base + C];
  float bl = img[base + (size_t)W * C];
  float br = img[base + (size_t)W * C + C];
  float top = tl + (tr - tl) * ax;
  float bot = bl + (br - bl) * ax;
  out[tid] = top + (bot - top) * ay;
}

// -------------------- 4x4 stride-2 transposed conv (SAME, pad 2,2) ----------
__global__ void convt_k(const float* __restrict__ s0, int c0,
                        const float* __restrict__ s1, int c1,
                        const float* __restrict__ w, const float* __restrict__ bias,
                        float* __restrict__ out, int NB, int Hi, int Wi, int Cout,
                        float scale) {
  int Ho = Hi * 2, Wo = Wi * 2;
  int Cin = c0 + c1;
  int n = NB * Ho * Wo * Cout;
  int tid = blockIdx.x * blockDim.x + threadIdx.x;
  if (tid >= n) return;
  int co = tid % Cout;
  int p = tid / Cout;
  int xo = p % Wo, yo = (p / Wo) % Ho, b = p / (Wo * Ho);
  float acc = bias[co];
  for (int kh = 0; kh < 4; kh++) {
    int d = yo + kh - 2;
    if (d < 0 || (d & 1)) continue;
    int yi = d >> 1;
    if (yi >= Hi) continue;
    for (int kw = 0; kw < 4; kw++) {
      int e = xo + kw - 2;
      if (e < 0 || (e & 1)) continue;
      int xi = e >> 1;
      if (xi >= Wi) continue;
      size_t base = (size_t)(b * Hi + yi) * Wi + xi;
      for (int ci = 0; ci < Cin; ci++) {
        float v = (ci < c0) ? s0[base * c0 + ci] : s1[base * c1 + (ci - c0)];
        acc += v * w[((size_t)(kh * 4 + kw) * Cin + ci) * Cout + co];
      }
    }
  }
  out[tid] = acc * scale;
}

// -------------------- residual add -----------------------------------------
__global__ void add_k(float* __restrict__ out, const float* __restrict__ a, int n) {
  int tid = blockIdx.x * blockDim.x + threadIdx.x;
  if (tid < n) out[tid] += a[tid];
}

// ===========================================================================
extern "C" void kernel_launch(void* const* d_in, const int* in_sizes, int n_in,
                              void* d_out, int out_size, void* d_ws, size_t ws_size,
                              hipStream_t stream) {
  (void)in_sizes; (void)n_in; (void)out_size; (void)ws_size;

  // ---- flattened input layout (setup_inputs dict order, leaves depth-first)
  // 0..4:  feature1_{2,3,4,5,6}   5..9: feature2_{2,3,4,5,6}
  // 10+12*i: dec{6,5,4,3,2} (cnv1..cnv6, each w,b)
  // 70..85: upflow3,upfeat3,upflow4,upfeat4,upflow5,upfeat5,upflow6,upfeat6
  // 86..99: cn cnv1..cnv7 (w,b)
  const int Hs[5] = {6, 12, 24, 48, 96};
  const int Ws[5] = {8, 16, 32, 64, 128};
  const int Cs[5] = {196, 128, 96, 64, 32};
  const size_t flowOff[5] = {0, 384, 1920, 8064, 32640};
  const float upflowScale[4] = {0.625f, 1.25f, 2.5f, 5.0f};

  float* dout = (float*)d_out;

  // ---- deterministic bump allocator over d_ws (~113 MB), sized for level 2
  float* wsf = (float*)d_ws;
  size_t off = 0;
  auto alloc = [&](size_t nfloats) -> float* {
    float* p = wsf + off;
    off += (nfloats + 63) & ~(size_t)63;
    return p;
  };
  const size_t MAXP = 4u * 96u * 128u;   // 49152
  float* cvb       = alloc(MAXP * 81);
  float* x1        = alloc(MAXP * 128);
  float* x2        = alloc(MAXP * 128);
  float* x3        = alloc(MAXP * 96);
  float* x4        = alloc(MAXP * 64);
  float* x5        = alloc(MAXP * 32);
  float* warpb     = alloc(MAXP * 32);   // max P*C over levels = 49152*32
  float* upflow_up = alloc(MAXP * 2);
  float* upfeat_up = alloc(MAXP * 2);
  float* cn_out    = alloc(MAXP * 2);
  unsigned short* packed = (unsigned short*)alloc(262144); // 512K bf16 >= max 393216

  auto run_conv = [&](const float* sA, int cA, const float* sB, int cB,
                      const float* sC, int cC, const float* sD, int cD,
                      const float* sE, int cE, int widx, int leaky, int dil,
                      int Cout, int H, int W, float* outp) {
    int CinTot = cA + cB + cC + cD + cE;
    int K = 9 * CinTot;
    int Ktiles = (K + 31) / 32;
    int Ntiles = (Cout + 15) / 16;
    int ngrp = (Ntiles + NTPW - 1) / NTPW;
    int NtilesPad = ngrp * NTPW;
    int P = 4 * H * W;
    int Mtiles = (P + 15) / 16;
    const float* wraw = (const float*)d_in[widx];
    const float* braw = (const float*)d_in[widx + 1];
    size_t ptotal = (size_t)NtilesPad * Ktiles * 512;
    int pblocks = (int)((ptotal + 255) / 256);
    if (pblocks > 4096) pblocks = 4096;
    if (pblocks < 1) pblocks = 1;
    pack_w_k<<<pblocks, 256, 0, stream>>>(wraw, packed, K, Cout, Ktiles, NtilesPad);
    long long waves = (long long)Mtiles * ngrp;
    int blocks = (int)((waves + 7) / 8);
    conv3x3_wmma_k<<<blocks, 256, 0, stream>>>(
        sA, cA, sB, cB, sC, cC, sD, cD, sE, cE, packed, braw, outp,
        4, H, W, Cout, dil, leaky, CinTot, K, Ktiles, ngrp, Mtiles);
  };

  for (int i = 0; i < 5; i++) {
    const int H = Hs[i], W = Ws[i], C = Cs[i];
    const int P = 4 * H * W;
    const float* f1 = (const float*)d_in[4 - i];
    const float* f2 = (const float*)d_in[9 - i];

    // warp f2 with upsampled flow (skip at coarsest level)
    const float* f2u = f2;
    if (i > 0) {
      int n = P * C;
      warp_k<<<(n + 255) / 256, 256, 0, stream>>>(f2, upflow_up, warpb, 4, H, W, C);
      f2u = warpb;
    }
    { int n = P * 81;
      costvol_k<<<(n + 255) / 256, 256, 0, stream>>>(f1, f2u, cvb, 4, H, W, C); }

    // ---- dense flow decoder
    const int db = 10 + i * 12;
    float* flowp = dout + flowOff[i];
    if (i == 0) {
      run_conv(cvb, 81, 0,0, 0,0, 0,0, 0,0,          db + 0,  1, 1, 128, H, W, x1);
      run_conv(cvb, 81, x1,128, 0,0, 0,0, 0,0,       db + 2,  1, 1, 128, H, W, x2);
    } else {
      run_conv(cvb, 81, f1, C, upflow_up, 2, upfeat_up, 2, 0,0,
                                                      db + 0,  1, 1, 128, H, W, x1);
      run_conv(cvb, 81, f1, C, upflow_up, 2, upfeat_up, 2, x1, 128,
                                                      db + 2,  1, 1, 128, H, W, x2);
    }
    run_conv(x1, 128, x2, 128, 0,0, 0,0, 0,0,        db + 4,  1, 1, 96, H, W, x3);
    run_conv(x2, 128, x3, 96,  0,0, 0,0, 0,0,        db + 6,  1, 1, 64, H, W, x4);
    run_conv(x3, 96,  x4, 64,  0,0, 0,0, 0,0,        db + 8,  1, 1, 32, H, W, x5);
    run_conv(x4, 64,  x5, 32,  0,0, 0,0, 0,0,        db + 10, 0, 1, 2,  H, W, flowp);

    // ---- upsample flow + feat to next level
    if (i < 4) {
      int n2 = 4 * (2 * H) * (2 * W) * 2;
      int wf = 82 - 4 * i, bf = 83 - 4 * i;   // upflow{6,5,4,3}
      convt_k<<<(n2 + 255) / 256, 256, 0, stream>>>(
          flowp, 2, (const float*)nullptr, 0,
          (const float*)d_in[wf], (const float*)d_in[bf],
          upflow_up, 4, H, W, 2, upflowScale[i]);
      int wu = 84 - 4 * i, bu = 85 - 4 * i;   // upfeat{6,5,4,3}
      convt_k<<<(n2 + 255) / 256, 256, 0, stream>>>(
          x4, 64, x5, 32,
          (const float*)d_in[wu], (const float*)d_in[bu],
          upfeat_up, 4, H, W, 2, 1.0f);
    }
  }

  // ---- context network at level 2 (input = concat(x4,x5) = 96ch), residual add
  {
    const int H = 96, W = 128;
    run_conv(x4, 64, x5, 32, 0,0, 0,0, 0,0, 86, 1, 1,  128, H, W, x1);
    run_conv(x1, 128, 0,0, 0,0, 0,0, 0,0,   88, 1, 2,  128, H, W, x2);
    run_conv(x2, 128, 0,0, 0,0, 0,0, 0,0,   90, 1, 4,  128, H, W, x1);
    run_conv(x1, 128, 0,0, 0,0, 0,0, 0,0,   92, 1, 8,  96,  H, W, x2);
    run_conv(x2, 96,  0,0, 0,0, 0,0, 0,0,   94, 1, 16, 64,  H, W, x1);
    run_conv(x1, 64,  0,0, 0,0, 0,0, 0,0,   96, 1, 1,  32,  H, W, x2);
    run_conv(x2, 32,  0,0, 0,0, 0,0, 0,0,   98, 0, 1,  2,   H, W, cn_out);
    add_k<<<(98304 + 255) / 256, 256, 0, stream>>>(dout + 32640, cn_out, 98304);
  }
}
